// GAT_skip_1_layer_default_7567732376259
// MI455X (gfx1250) — matrix operand order repaired and checked by reading, not compiled, not run on hardware
//
#include <hip/hip_runtime.h>
#include <hip/hip_bf16.h>

typedef __attribute__((ext_vector_type(16))) _Float16 v16h;
typedef __attribute__((ext_vector_type(8)))  float    v8f;

#define GN    50000
#define GE    800000
#define GET   (GE + GN)          // edges + self loops
#define FEAT  128
#define HEADS 4
#define HID   32
#define OUTC  40
#define ROWT  (GN / 16)          // 3125 row tiles, exact
#define NEGS  0.2f
#define BNEPS 1e-5f
#define SKIPC 0.5f

// ---------------------------------------------------------------- utilities
__global__ void fill_zero(float* __restrict__ p, long long n) {
    long long i = (long long)blockIdx.x * blockDim.x + threadIdx.x;
    long long s = (long long)gridDim.x * blockDim.x;
    for (; i < n; i += s) p[i] = 0.0f;
}

// ---------------------------------------------------------------- WMMA GEMM
// O[m, j] = sum_k X[m,k] * W[j,k]   (X: [M x 128] rowmajor, W: [ncols x 128])
// One wave per 16x16 output tile. K = 128 -> 4 x v_wmma_f32_16x16x32_f16.
// For the ragged last column tile (layer 2, ncols=40) we CLAMP the B row
// pointer instead of zero-masking data: WMMA column N of D depends only on
// column N of B, held by exactly the lanes whose stores are masked below,
// so clamped-garbage columns never pollute valid output.
__global__ __launch_bounds__(32) void wmma_gemm_f16(
    const float* __restrict__ X, const float* __restrict__ W,
    float* __restrict__ O, int ldo, int ncols)
{
    const int lane = threadIdx.x & 31;
    const int mt = blockIdx.x;
    const int nt = blockIdx.y;
    const int row = mt * 16 + (lane & 15);
    const int col = nt * 16 + (lane & 15);
    const bool colValid = (col < ncols);
    const int colc = colValid ? col : (ncols - 1);   // in-bounds clamp

    const int aK0 = (lane < 16) ? 0 : 16;   // contiguous k: lane half h holds k 16h..16h+15
    const int bK0 = (lane < 16) ? 0 : 16;   // contiguous k: lane half h holds k 16h..16h+15

    const float* arow = X + (size_t)row  * FEAT + aK0;
    const float* brow = W + (size_t)colc * FEAT + bK0;

    v8f c = {};
    #pragma unroll
    for (int kt = 0; kt < 4; ++kt) {
        const float* ap = arow + kt * 32;
        const float* bp = brow + kt * 32;

        v16h a, b;
        #pragma unroll
        for (int i = 0; i < 8; ++i)  a[i]     = (_Float16)ap[i];
        #pragma unroll
        for (int i = 0; i < 8; ++i)  a[8 + i] = (_Float16)ap[8 + i];
        #pragma unroll
        for (int i = 0; i < 16; ++i) b[i] = (_Float16)bp[i];

        c = __builtin_amdgcn_wmma_f32_16x16x32_f16(
                false, a, false, b, (short)0, c, false, false);
    }
    if (colValid) {
        const int rbase = mt * 16 + ((lane < 16) ? 0 : 8);
        #pragma unroll
        for (int r = 0; r < 8; ++r)
            O[(size_t)(rbase + r) * ldo + col] = c[r];
    }
}

// ------------------------------------------------------- attention logits
__global__ void att_logits_h4(const float* __restrict__ h,
                              const float* __restrict__ a_s,
                              const float* __restrict__ a_d,
                              float* __restrict__ als, float* __restrict__ ald)
{
    int t = blockIdx.x * blockDim.x + threadIdx.x;
    if (t >= GN * HEADS) return;
    int n = t >> 2, hh = t & 3;
    const float* hp = h + (size_t)n * FEAT + hh * HID;
    const float* sp = a_s + hh * HID;
    const float* dp = a_d + hh * HID;
    float s = 0.f, d = 0.f;
    #pragma unroll
    for (int i = 0; i < HID; ++i) { float v = hp[i]; s += v * sp[i]; d += v * dp[i]; }
    als[t] = s; ald[t] = d;
}

__global__ void att_logits_h1(const float* __restrict__ h,
                              const float* __restrict__ a_s,
                              const float* __restrict__ a_d,
                              float* __restrict__ als, float* __restrict__ ald)
{
    int n = blockIdx.x * blockDim.x + threadIdx.x;
    if (n >= GN) return;
    const float* hp = h + (size_t)n * OUTC;
    float s = 0.f, d = 0.f;
    #pragma unroll
    for (int i = 0; i < OUTC; ++i) { float v = hp[i]; s += v * a_s[i]; d += v * a_d[i]; }
    als[n] = s; ald[n] = d;
}

// -------------------------------------------------- edge softmax aggregate
// One wave per edge. Softmax shift dropped (exp-safe logits); denom tracked.
__global__ void edge_agg_h4(const float* __restrict__ h,
                            const float* __restrict__ als,
                            const float* __restrict__ ald,
                            const long long* __restrict__ srcI,
                            const long long* __restrict__ dstI,
                            float* __restrict__ acc, float* __restrict__ denom)
{
    int wid  = (blockIdx.x * blockDim.x + threadIdx.x) >> 5;
    int lane = threadIdx.x & 31;
    if (wid >= GET) return;
    int s, d;
    if (wid < GE) { s = (int)srcI[wid]; d = (int)dstI[wid]; }
    else          { s = d = wid - GE; }

    float w = 0.f;
    if (lane < HEADS) {
        float e = als[s * HEADS + lane] + ald[d * HEADS + lane];
        e = (e > 0.f) ? e : NEGS * e;
        w = __expf(e);
        atomicAdd(&denom[d * HEADS + lane], w);
    }
    float w0 = __shfl(w, 0), w1 = __shfl(w, 1), w2 = __shfl(w, 2), w3 = __shfl(w, 3);
    int c0 = lane * 4;                       // 4 channels per lane, same head
    float ww = (lane < 8) ? w0 : (lane < 16) ? w1 : (lane < 24) ? w2 : w3;
    const float4 hv = *(const float4*)(h + (size_t)s * FEAT + c0);
    float* a = acc + (size_t)d * FEAT + c0;
    atomicAdd(a + 0, ww * hv.x);
    atomicAdd(a + 1, ww * hv.y);
    atomicAdd(a + 2, ww * hv.z);
    atomicAdd(a + 3, ww * hv.w);
}

__global__ void edge_agg_h1(const float* __restrict__ h,
                            const float* __restrict__ als,
                            const float* __restrict__ ald,
                            const long long* __restrict__ srcI,
                            const long long* __restrict__ dstI,
                            float* __restrict__ acc, float* __restrict__ denom)
{
    int wid  = (blockIdx.x * blockDim.x + threadIdx.x) >> 5;
    int lane = threadIdx.x & 31;
    if (wid >= GET) return;
    int s, d;
    if (wid < GE) { s = (int)srcI[wid]; d = (int)dstI[wid]; }
    else          { s = d = wid - GE; }

    float w = 0.f;
    if (lane == 0) {
        float e = als[s] + ald[d];
        e = (e > 0.f) ? e : NEGS * e;
        w = __expf(e);
        atomicAdd(&denom[d], w);
    }
    w = __shfl(w, 0);
    const float* hp = h + (size_t)s * OUTC;
    float* a = acc + (size_t)d * OUTC;
    atomicAdd(a + lane, w * hp[lane]);
    if (lane < OUTC - 32) atomicAdd(a + 32 + lane, w * hp[32 + lane]);
}

// ---------------------------------------------------------------- finalize
__global__ void finalize_h4(const float* __restrict__ acc,
                            const float* __restrict__ denom,
                            const float* __restrict__ bias,
                            float* __restrict__ z)
{
    long long idx = (long long)blockIdx.x * blockDim.x + threadIdx.x;
    if (idx >= (long long)GN * FEAT) return;
    int c = (int)(idx & (FEAT - 1));
    long long n = idx >> 7;
    z[idx] = acc[idx] / denom[n * HEADS + (c >> 5)] + bias[c];
}

__global__ void finalize_out(const float* __restrict__ acc,
                             const float* __restrict__ denom,
                             const float* __restrict__ bias,
                             float* __restrict__ out)
{
    long long idx = (long long)blockIdx.x * blockDim.x + threadIdx.x;
    if (idx >= (long long)GN * OUTC) return;
    int n = (int)(idx / OUTC), c = (int)(idx % OUTC);
    out[idx] = acc[idx] / denom[n] + bias[c];
}

// --------------------------------------------------------------- batchnorm
__global__ void bn_stats(const float* __restrict__ z,
                         float* __restrict__ sum, float* __restrict__ sq)
{
    int c = threadIdx.x;                 // blockDim.x == 128
    float s = 0.f, q = 0.f;
    for (int r = blockIdx.x; r < GN; r += gridDim.x) {
        float v = z[(size_t)r * FEAT + c];
        s += v; q += v * v;
    }
    atomicAdd(&sum[c], s);
    atomicAdd(&sq[c], q);
}

__global__ void bn_apply(const float* __restrict__ z,
                         const float* __restrict__ sum, const float* __restrict__ sq,
                         const float* __restrict__ g, const float* __restrict__ be,
                         const float* __restrict__ skip, float* __restrict__ out)
{
    long long idx = (long long)blockIdx.x * blockDim.x + threadIdx.x;
    if (idx >= (long long)GN * FEAT) return;
    int c = (int)(idx & (FEAT - 1));
    const float invN = 1.0f / (float)GN;
    float mu  = sum[c] * invN;
    float var = sq[c] * invN - mu * mu;
    float v = (z[idx] - mu) * rsqrtf(var + BNEPS) * g[c] + be[c];
    if (skip) v += SKIPC * skip[idx];
    out[idx] = fmaxf(v, 0.0f);
}

// ------------------------------------------------------------------ launch
extern "C" void kernel_launch(void* const* d_in, const int* in_sizes, int n_in,
                              void* d_out, int out_size, void* d_ws, size_t ws_size,
                              hipStream_t stream)
{
    const float* x   = (const float*)d_in[0];
    const float* W0  = (const float*)d_in[1];
    const float* as0 = (const float*)d_in[2];
    const float* ad0 = (const float*)d_in[3];
    const float* b0  = (const float*)d_in[4];
    const float* g0  = (const float*)d_in[5];
    const float* be0 = (const float*)d_in[6];
    const float* W1  = (const float*)d_in[7];
    const float* as1 = (const float*)d_in[8];
    const float* ad1 = (const float*)d_in[9];
    const float* b1  = (const float*)d_in[10];
    const float* g1  = (const float*)d_in[11];
    const float* be1 = (const float*)d_in[12];
    const float* W2  = (const float*)d_in[13];
    const float* as2 = (const float*)d_in[14];
    const float* ad2 = (const float*)d_in[15];
    const float* b2  = (const float*)d_in[16];
    const long long* ei = (const long long*)d_in[17];
    const long long* srcI = ei;
    const long long* dstI = ei + GE;
    float* out = (float*)d_out;

    float* ws = (float*)d_ws;
    const size_t NF = (size_t)GN * FEAT;
    float* bufA   = ws;              // h / z scratch        [N*128]
    float* bufB   = bufA + NF;       // aggregation accum    [N*128]
    float* bufC   = bufB + NF;       // post-BN activations  [N*128]
    float* als    = bufC + NF;       // [N*4]
    float* ald    = als + (size_t)GN * HEADS;
    float* denom  = ald + (size_t)GN * HEADS;
    float* bnsum  = denom + (size_t)GN * HEADS;  // [128]
    float* bnsq   = bnsum + FEAT;                // [128]
    float* als2   = bnsq + FEAT;                 // [N]
    float* ald2   = als2 + GN;
    float* denom2 = ald2 + GN;

    const dim3 blk256(256), blk128(128), blk32(32);
    const dim3 gGemm(ROWT, FEAT / 16);           // 3125 x 8
    const dim3 gGemm2(ROWT, 3);                  // 40 cols -> 3 tiles
    const int  gLog4 = (GN * HEADS + 255) / 256;
    const int  gLog1 = (GN + 255) / 256;
    const int  gEdge = (GET * 32 + 255) / 256;
    const int  gElem = (int)((NF + 255) / 256);
    const int  gOut  = (GN * OUTC + 255) / 256;

    // ---------------- Layer 0 ----------------
    wmma_gemm_f16<<<gGemm, blk32, 0, stream>>>(x, W0, bufA, FEAT, FEAT);
    att_logits_h4<<<gLog4, blk256, 0, stream>>>(bufA, as0, ad0, als, ald);
    fill_zero<<<2048, blk256, 0, stream>>>(bufB, (long long)NF);
    fill_zero<<<256, blk256, 0, stream>>>(denom, (long long)GN * HEADS);
    edge_agg_h4<<<gEdge, blk256, 0, stream>>>(bufA, als, ald, srcI, dstI, bufB, denom);
    finalize_h4<<<gElem, blk256, 0, stream>>>(bufB, denom, b0, bufA);
    fill_zero<<<1, blk256, 0, stream>>>(bnsum, 2 * FEAT);   // bnsum + bnsq contiguous
    bn_stats<<<512, blk128, 0, stream>>>(bufA, bnsum, bnsq);
    bn_apply<<<gElem, blk256, 0, stream>>>(bufA, bnsum, bnsq, g0, be0, nullptr, bufC);

    // ---------------- Layer 1 ----------------
    wmma_gemm_f16<<<gGemm, blk32, 0, stream>>>(bufC, W1, bufA, FEAT, FEAT);
    att_logits_h4<<<gLog4, blk256, 0, stream>>>(bufA, as1, ad1, als, ald);
    fill_zero<<<2048, blk256, 0, stream>>>(bufB, (long long)NF);
    fill_zero<<<256, blk256, 0, stream>>>(denom, (long long)GN * HEADS);
    edge_agg_h4<<<gEdge, blk256, 0, stream>>>(bufA, als, ald, srcI, dstI, bufB, denom);
    finalize_h4<<<gElem, blk256, 0, stream>>>(bufB, denom, b1, bufA);
    fill_zero<<<1, blk256, 0, stream>>>(bnsum, 2 * FEAT);
    bn_stats<<<512, blk128, 0, stream>>>(bufA, bnsum, bnsq);
    bn_apply<<<gElem, blk256, 0, stream>>>(bufA, bnsum, bnsq, g1, be1, bufC, bufC);

    // ---------------- Layer 2 ----------------
    wmma_gemm_f16<<<gGemm2, blk32, 0, stream>>>(bufC, W2, bufB, OUTC, OUTC);
    att_logits_h1<<<gLog1, blk256, 0, stream>>>(bufB, as2, ad2, als2, ald2);
    fill_zero<<<1024, blk256, 0, stream>>>(bufA, (long long)GN * OUTC);
    fill_zero<<<128, blk256, 0, stream>>>(denom2, (long long)GN);
    edge_agg_h1<<<gEdge, blk256, 0, stream>>>(bufB, als2, ald2, srcI, dstI, bufA, denom2);
    finalize_out<<<gOut, blk256, 0, stream>>>(bufA, denom2, b2, out);
}
